// GAT_27419071218300
// MI455X (gfx1250) — matrix operand order repaired and checked by reading, not compiled, not run on hardware
//
#include <hip/hip_runtime.h>
#include <hip/hip_bf16.h>
#include <math.h>

typedef __attribute__((ext_vector_type(16))) _Float16 v16h;
typedef __attribute__((ext_vector_type(8)))  float    v8f;
typedef __attribute__((ext_vector_type(4)))  float    v4f;

#define FIN   256   // input feature dim
#define H1N   8     // layer1 heads
#define C1N   8     // layer1 channels/head
#define D1    64    // H1*C1
#define C2N   40    // layer2 classes
#define D2PAD 48    // 40 padded to 3x16 WMMA tiles

// ---------- ordered-float <-> uint monotone transform (for atomic max) ----------
static __device__ __forceinline__ unsigned f2ord(float f) {
  unsigned u = __float_as_uint(f);
  return (u & 0x80000000u) ? ~u : (u | 0x80000000u);
}
static __device__ __forceinline__ float ord2f(unsigned u) {
  return __uint_as_float((u & 0x80000000u) ? (u & 0x7fffffffu) : ~u);
}
static inline unsigned f2ord_host(float f) {
  union { float f; unsigned u; } c; c.f = f;
  return (c.u & 0x80000000u) ? ~c.u : (c.u | 0x80000000u);
}

// ---------- fill kernels ----------
__global__ void fill_f32_k(float* p, float v, long long n) {
  long long i = blockIdx.x * (long long)blockDim.x + threadIdx.x;
  if (i < n) p[i] = v;
}
__global__ void fill_u32_k(unsigned* p, unsigned v, long long n) {
  long long i = blockIdx.x * (long long)blockDim.x + threadIdx.x;
  if (i < n) p[i] = v;
}

// ---------- GEMM1: h1[N,64] = x[N,256] @ W1[256,64] (WMMA f16->f32) ----------
// block = 256 threads = 8 waves; wave w computes rows [row0+16w, row0+16w+16)
// sB is fragment-swizzled: [chunk(8)][nt(4)][lane(32)][idx(16)] halves so a
// B fragment is one 32B-aligned v16h load (2x ds_load_b128).
__global__ __launch_bounds__(256) void gemm1_k(const float* __restrict__ x,
                                               const float* __restrict__ W1,
                                               float* __restrict__ h1, int n) {
  __shared__ _Float16 sB[FIN * D1];   // 32 KB
  __shared__ _Float16 sA[128 * 32];   // 8 KB
  const int tid  = threadIdx.x;
  const int lane = tid & 31;
  const int wv   = tid >> 5;
  const int hi   = lane >> 4;
  const int m    = lane & 15;
  const int row0 = blockIdx.x * 128;

  // stage W1 (coalesced global reads, swizzled LDS writes; once per block)
  for (int i = tid; i < FIN * D1; i += 256) {
    int k = i >> 6, c = i & 63;           // W1 row-major [256,64]
    int chunk = k >> 5, kl = k & 31;
    int dl = ((kl >> 4) << 4) | (c & 15); // dest lane within fragment
    int dest = (((chunk << 2) | (c >> 4)) * 32 + dl) * 16 + (kl & 15);
    sB[dest] = (_Float16)W1[i];
  }

  v8f acc[4] = {};
  for (int kk = 0; kk < FIN; kk += 32) {
    __syncthreads();
    #pragma unroll
    for (int j = 0; j < 16; ++j) {        // stage 128x32 A chunk, coalesced
      int i = tid + 256 * j;
      int r = i >> 5, c = i & 31;
      int rr = row0 + r; if (rr >= n) rr = n - 1;
      sA[i] = (_Float16)x[(long long)rr * FIN + kk + c];
    }
    __syncthreads();
    // A fragment: contiguous k = 16*hi + idx (matches sB swizzle) -> two b128 LDS loads
    v16h a;
    #pragma unroll
    for (int idx = 0; idx < 16; ++idx) {
      int kl = idx + (hi << 4);
      a[idx] = sA[(wv * 16 + m) * 32 + kl];
    }
    const int chunk = kk >> 5;
    #pragma unroll
    for (int nt = 0; nt < 4; ++nt) {
      const v16h b = *reinterpret_cast<const v16h*>(
          &sB[(((chunk << 2) | nt) * 32 + lane) * 16]);
      acc[nt] = __builtin_amdgcn_wmma_f32_16x16x32_f16(
          false, a, false, b, (short)0, acc[nt], false, false);
    }
  }
  // C/D layout: VGPR r -> M = r + 8*hi, N = m
  if (row0 + 128 <= n) {                  // block-uniform fast path
    float* base = h1 + (long long)(row0 + wv * 16 + (hi << 3)) * D1 + m;
    #pragma unroll
    for (int nt = 0; nt < 4; ++nt)
      #pragma unroll
      for (int r = 0; r < 8; ++r)
        base[r * D1 + nt * 16] = acc[nt][r];
  } else {
    #pragma unroll
    for (int nt = 0; nt < 4; ++nt)
      #pragma unroll
      for (int r = 0; r < 8; ++r) {
        int row = row0 + wv * 16 + r + (hi << 3);
        if (row < n) h1[(long long)row * D1 + nt * 16 + m] = acc[nt][r];
      }
  }
}

// ---------- GEMM2: h2[N,40] = h1p[N,64] @ W2[64,40] (padded to 48 cols) ----------
__global__ __launch_bounds__(256) void gemm2_k(const float* __restrict__ hin,
                                               const float* __restrict__ W2,
                                               float* __restrict__ h2, int n) {
  __shared__ _Float16 sB[D1 * D2PAD]; // swizzled [chunk(2)][nt(3)][lane][idx]
  __shared__ _Float16 sA[128 * 32];
  const int tid  = threadIdx.x;
  const int lane = tid & 31;
  const int wv   = tid >> 5;
  const int hi   = lane >> 4;
  const int m    = lane & 15;
  const int row0 = blockIdx.x * 128;

  for (int i = tid; i < D1 * D2PAD; i += 256) {
    int k = i / D2PAD, c = i % D2PAD;
    float v = (c < C2N) ? W2[k * C2N + c] : 0.0f;
    int chunk = k >> 5, kl = k & 31;
    int dl = ((kl >> 4) << 4) | (c & 15);
    int dest = ((chunk * 3 + (c >> 4)) * 32 + dl) * 16 + (kl & 15);
    sB[dest] = (_Float16)v;
  }

  v8f acc[3] = {};
  for (int kk = 0; kk < D1; kk += 32) {
    __syncthreads();
    #pragma unroll
    for (int j = 0; j < 16; ++j) {
      int i = tid + 256 * j;
      int r = i >> 5, c = i & 31;
      int rr = row0 + r; if (rr >= n) rr = n - 1;
      sA[i] = (_Float16)hin[(long long)rr * D1 + kk + c];
    }
    __syncthreads();
    v16h a;
    #pragma unroll
    for (int idx = 0; idx < 16; ++idx) {
      int kl = idx + (hi << 4);
      a[idx] = sA[(wv * 16 + m) * 32 + kl];
    }
    const int chunk = kk >> 5;
    #pragma unroll
    for (int nt = 0; nt < 3; ++nt) {
      const v16h b = *reinterpret_cast<const v16h*>(
          &sB[((chunk * 3 + nt) * 32 + lane) * 16]);
      acc[nt] = __builtin_amdgcn_wmma_f32_16x16x32_f16(
          false, a, false, b, (short)0, acc[nt], false, false);
    }
  }
  if (row0 + 128 <= n) {
    long long rbase = (long long)(row0 + wv * 16 + (hi << 3)) * C2N + m;
    #pragma unroll
    for (int nt = 0; nt < 2; ++nt)        // cols 0..31 always valid
      #pragma unroll
      for (int r = 0; r < 8; ++r)
        h2[rbase + r * C2N + nt * 16] = acc[nt][r];
    if (m < 8) {                          // cols 32..39 from third tile
      #pragma unroll
      for (int r = 0; r < 8; ++r)
        h2[rbase + r * C2N + 32] = acc[2][r];
    }
  } else {
    #pragma unroll
    for (int nt = 0; nt < 3; ++nt)
      #pragma unroll
      for (int r = 0; r < 8; ++r) {
        int row = row0 + wv * 16 + r + (hi << 3);
        int col = nt * 16 + m;
        if (row < n && col < C2N) h2[(long long)row * C2N + col] = acc[nt][r];
      }
  }
}

// ---------- attention logits per node: a_src/a_dst [N,H] ----------
__global__ void attn_k(const float* __restrict__ h, const float* __restrict__ att_s,
                       const float* __restrict__ att_d, float* __restrict__ as,
                       float* __restrict__ ad, int n, int H, int C) {
  int i = blockIdx.x * blockDim.x + threadIdx.x;
  if (i >= n * H) return;
  int node = i / H, hh = i % H;
  const float* hp = h + (long long)node * H * C + hh * C;
  float s = 0.f, d = 0.f;
  for (int c = 0; c < C; ++c) {
    float v = hp[c];
    s += v * att_s[hh * C + c];
    d += v * att_d[hh * C + c];
  }
  as[i] = s; ad[i] = d;
}

// ---------- edge helpers ----------
static __device__ __forceinline__ float edge_logit(const int* srcA, const int* dstA,
                                                   int E, int e, int H, int hh,
                                                   const float* as, const float* ad,
                                                   int& s, int& d) {
  if (e < E) { s = srcA[e]; d = dstA[e]; } else { s = d = e - E; }
  float v = as[s * H + hh] + ad[d * H + hh];
  return v > 0.f ? v : 0.2f * v;  // leaky_relu(0.2)
}

__global__ void edge_max_k(const int* __restrict__ srcA, const int* __restrict__ dstA,
                           int E, int Et, const float* __restrict__ as,
                           const float* __restrict__ ad, unsigned* __restrict__ emax, int H) {
  long long i = blockIdx.x * (long long)blockDim.x + threadIdx.x;
  if (i >= (long long)Et * H) return;
  int e = (int)(i / H), hh = (int)(i % H), s, d;
  float v = edge_logit(srcA, dstA, E, e, H, hh, as, ad, s, d);
  atomicMax(&emax[d * H + hh], f2ord(v));
}

__global__ void edge_sum_k(const int* __restrict__ srcA, const int* __restrict__ dstA,
                           int E, int Et, const float* __restrict__ as,
                           const float* __restrict__ ad, const unsigned* __restrict__ emax,
                           float* __restrict__ denom, int H) {
  long long i = blockIdx.x * (long long)blockDim.x + threadIdx.x;
  if (i >= (long long)Et * H) return;
  int e = (int)(i / H), hh = (int)(i % H), s, d;
  float v = edge_logit(srcA, dstA, E, e, H, hh, as, ad, s, d);
  float w = expf(v - ord2f(emax[d * H + hh]));
  atomicAdd(&denom[d * H + hh], w);
}

__global__ void edge_msg_k(const int* __restrict__ srcA, const int* __restrict__ dstA,
                           int E, int Et, const float* __restrict__ as,
                           const float* __restrict__ ad, const unsigned* __restrict__ emax,
                           const float* __restrict__ denom, const float* __restrict__ feat,
                           float* __restrict__ out, int H, int C) {
  long long i = blockIdx.x * (long long)blockDim.x + threadIdx.x;
  if (i >= (long long)Et * H) return;
  int e = (int)(i / H), hh = (int)(i % H), s, d;
  float v = edge_logit(srcA, dstA, E, e, H, hh, as, ad, s, d);
  float alpha = expf(v - ord2f(emax[d * H + hh])) / (denom[d * H + hh] + 1e-16f);
  // head offsets are multiples of 32B; row strides 256B (L1) / 160B (L2):
  // feat/out sub-rows are 16B-aligned -> vector gathers are safe.
  const v4f* fp4 = (const v4f*)(feat + (long long)s * H * C + hh * C);
  float* op = out + (long long)d * H * C + hh * C;
  int C4 = C >> 2;   // C is 8 or 40
  for (int q = 0; q < C4; ++q) {
    v4f f = fp4[q];
    atomicAdd(op + 4 * q + 0, f[0] * alpha);
    atomicAdd(op + 4 * q + 1, f[1] * alpha);
    atomicAdd(op + 4 * q + 2, f[2] * alpha);
    atomicAdd(op + 4 * q + 3, f[3] * alpha);
  }
}

// ---------- bias + ELU, in place ----------
__global__ void elu_bias_k(float* __restrict__ a, const float* __restrict__ bias,
                           long long n, int K) {
  long long i = blockIdx.x * (long long)blockDim.x + threadIdx.x;
  if (i >= n) return;
  float v = a[i] + bias[(int)(i % K)];
  a[i] = v > 0.f ? v : (expf(v) - 1.0f);
}

// ---------- bias + log_softmax over 40 classes, in place on d_out ----------
__global__ void lsm_k(float* __restrict__ out, const float* __restrict__ bias, int n) {
  int i = blockIdx.x * blockDim.x + threadIdx.x;
  if (i >= n) return;
  float v[C2N];
  float* row = out + (long long)i * C2N;
  float mx = -3.0e38f;
  for (int c = 0; c < C2N; ++c) { v[c] = row[c] + bias[c]; mx = fmaxf(mx, v[c]); }
  float sum = 0.f;
  for (int c = 0; c < C2N; ++c) sum += expf(v[c] - mx);
  float lse = mx + logf(sum);
  for (int c = 0; c < C2N; ++c) row[c] = v[c] - lse;
}

extern "C" void kernel_launch(void* const* d_in, const int* in_sizes, int n_in,
                              void* d_out, int out_size, void* d_ws, size_t ws_size,
                              hipStream_t stream) {
  (void)n_in; (void)out_size; (void)ws_size;
  const float* x        = (const float*)d_in[0];
  const int*   eidx     = (const int*)d_in[1];
  const float* W1       = (const float*)d_in[2];
  const float* att_src1 = (const float*)d_in[3];
  const float* att_dst1 = (const float*)d_in[4];
  const float* bias1    = (const float*)d_in[5];
  const float* W2       = (const float*)d_in[6];
  const float* att_src2 = (const float*)d_in[7];
  const float* att_dst2 = (const float*)d_in[8];
  const float* bias2    = (const float*)d_in[9];

  const int N  = in_sizes[0] / FIN;   // 100000
  const int E  = in_sizes[1] / 2;     // 1600000
  const int Et = E + N;               // + self loops
  const int* srcA = eidx;
  const int* dstA = eidx + E;

  char* wsp = (char*)d_ws;
  size_t off = 0;
  auto alloc = [&](size_t bytes) -> void* {
    void* p = wsp + off; off += (bytes + 255) & ~(size_t)255; return p;
  };
  float*    h1   = (float*)alloc((size_t)N * D1 * 4);
  float*    as1  = (float*)alloc((size_t)N * H1N * 4);
  float*    ad1  = (float*)alloc((size_t)N * H1N * 4);
  unsigned* em1  = (unsigned*)alloc((size_t)N * H1N * 4);
  float*    dn1  = (float*)alloc((size_t)N * H1N * 4);
  float*    acc1 = (float*)alloc((size_t)N * D1 * 4);   // message accum -> h1p after ELU
  float*    h2   = (float*)alloc((size_t)N * C2N * 4);
  float*    as2  = (float*)alloc((size_t)N * 4);
  float*    ad2  = (float*)alloc((size_t)N * 4);
  unsigned* em2  = (unsigned*)alloc((size_t)N * 4);
  float*    dn2  = (float*)alloc((size_t)N * 4);
  float*    oacc = (float*)d_out;                        // layer-2 accum in place

  const int TPB = 256;
  const unsigned NEG_ORD = f2ord_host(-3.0e38f);
  auto gr = [&](long long nn) { return (int)((nn + TPB - 1) / TPB); };

  // ---- layer 1 ----
  gemm1_k<<<(N + 127) / 128, 256, 0, stream>>>(x, W1, h1, N);
  attn_k<<<gr((long long)N * H1N), TPB, 0, stream>>>(h1, att_src1, att_dst1, as1, ad1, N, H1N, C1N);
  fill_u32_k<<<gr((long long)N * H1N), TPB, 0, stream>>>(em1, NEG_ORD, (long long)N * H1N);
  fill_f32_k<<<gr((long long)N * H1N), TPB, 0, stream>>>(dn1, 0.f, (long long)N * H1N);
  fill_f32_k<<<gr((long long)N * D1), TPB, 0, stream>>>(acc1, 0.f, (long long)N * D1);
  long long t1 = (long long)Et * H1N;
  edge_max_k<<<gr(t1), TPB, 0, stream>>>(srcA, dstA, E, Et, as1, ad1, em1, H1N);
  edge_sum_k<<<gr(t1), TPB, 0, stream>>>(srcA, dstA, E, Et, as1, ad1, em1, dn1, H1N);
  edge_msg_k<<<gr(t1), TPB, 0, stream>>>(srcA, dstA, E, Et, as1, ad1, em1, dn1, h1, acc1, H1N, C1N);
  elu_bias_k<<<gr((long long)N * D1), TPB, 0, stream>>>(acc1, bias1, (long long)N * D1, D1);

  // ---- layer 2 ----
  gemm2_k<<<(N + 127) / 128, 256, 0, stream>>>(acc1, W2, h2, N);
  attn_k<<<gr(N), TPB, 0, stream>>>(h2, att_src2, att_dst2, as2, ad2, N, 1, C2N);
  fill_u32_k<<<gr(N), TPB, 0, stream>>>(em2, NEG_ORD, N);
  fill_f32_k<<<gr(N), TPB, 0, stream>>>(dn2, 0.f, N);
  fill_f32_k<<<gr((long long)N * C2N), TPB, 0, stream>>>(oacc, 0.f, (long long)N * C2N);
  long long t2 = (long long)Et;
  edge_max_k<<<gr(t2), TPB, 0, stream>>>(srcA, dstA, E, Et, as2, ad2, em2, 1);
  edge_sum_k<<<gr(t2), TPB, 0, stream>>>(srcA, dstA, E, Et, as2, ad2, em2, dn2, 1);
  edge_msg_k<<<gr(t2), TPB, 0, stream>>>(srcA, dstA, E, Et, as2, ad2, em2, dn2, h2, oacc, 1, C2N);
  lsm_k<<<gr(N), TPB, 0, stream>>>(oacc, bias2, N);
}
